// Module_54631984005189
// MI455X (gfx1250) — matrix operand order repaired and checked by reading, not compiled, not run on hardware
//
#include <hip/hip_runtime.h>
#include <hip/hip_bf16.h>

// Problem constants (match reference)
#define BSZ   32
#define LSEQ  8192
#define DIM   256
#define NH    8
#define HD    32
#define CHUNK 128
#define NCHUNK (LSEQ / CHUNK)   // 64
#define SCALE  0.17677669529663687f  // 1/sqrt(32)

typedef __attribute__((ext_vector_type(16))) __bf16 v16bf;
typedef __attribute__((ext_vector_type(8)))  float  v8f;
typedef __attribute__((ext_vector_type(4)))  float  vf4;   // native vector for NT loads

union FragBF { v16bf v; uint4 q[2]; unsigned short s[16]; };

__device__ __forceinline__ unsigned short f32_to_bf16_rn(float f) {
  unsigned int u = __builtin_bit_cast(unsigned int, f);
  u += 0x7FFFu + ((u >> 16) & 1u);
  return (unsigned short)(u >> 16);
}

// ---- workspace layout (floats) ----
// [0,8192)        Qp        [B][D]
// [8192,8448)     cvals     [B][H]    (scale * Qp_h . bk_h)
// [8448,8704)     denom     [B][H]
// [8704,74240)    U         [B][H][D]
// [74240,139776)  qt (bf16) [B][16][D]  rows 8..15 zero-padded (131072 halves)
#define WS_QP   0
#define WS_C    8192
#define WS_DEN  8448
#define WS_U    8704
#define WS_QT   74240
#define QT_ROWS 16              // 8 real heads + 8 zero rows (WMMA N-pad)

// LDS strides (halves); padded to de-conflict banks
#define KSTR 264   // K chunk row stride (128 rows x 256 cols bf16)
#define VSTR 136   // V^T row stride (256 cols x 128 rows bf16)

// ---------- k0: Qp = Q@Wq.T + bq ; zero U, denom ----------
__global__ void k0_prep(const float* __restrict__ Q, const float* __restrict__ Wq,
                        const float* __restrict__ bq, float* __restrict__ ws) {
  int b = blockIdx.x, j = threadIdx.x;
  const float* q = Q + (size_t)b * DIM;
  const float* w = Wq + (size_t)j * DIM;
  float s = bq[j];
  #pragma unroll 4
  for (int k = 0; k < DIM; ++k) s += q[k] * w[k];
  ws[WS_QP + b * DIM + j] = s;
  float* U = ws + WS_U + (size_t)b * NH * DIM;
  #pragma unroll
  for (int i = 0; i < 8; ++i) U[j * 8 + i] = 0.0f;
  if (b == 0) ws[WS_DEN + j] = 0.0f;
}

// ---------- k1: qt[b,h,k] = scale * Qp_h . Wk[h*32+.., k]; zero pad rows; cvals ----------
__global__ void k1_qt(const float* __restrict__ Wk, const float* __restrict__ bk,
                      float* __restrict__ ws) {
  int b = blockIdx.x, t = threadIdx.x;   // t = output column k
  const float* Qp = ws + WS_QP + (size_t)b * DIM;
  unsigned short* qt = (unsigned short*)(ws + WS_QT) + (size_t)b * QT_ROWS * DIM;
  #pragma unroll
  for (int h = 0; h < NH; ++h) {
    float s = 0.0f;
    const float* qp = Qp + h * HD;
    #pragma unroll 8
    for (int hd = 0; hd < HD; ++hd) s += qp[hd] * Wk[(size_t)(h * HD + hd) * DIM + t];
    qt[h * DIM + t] = f32_to_bf16_rn(s * SCALE);
  }
  #pragma unroll
  for (int h = NH; h < QT_ROWS; ++h) qt[h * DIM + t] = 0;   // N-pad rows
  if (t < NH) {
    float s = 0.0f;
    const float* qp = Qp + t * HD;
    #pragma unroll 8
    for (int hd = 0; hd < HD; ++hd) s += qp[hd] * bk[t * HD + hd];
    ws[WS_C + b * NH + t] = s * SCALE;
  }
}

// ---------- k2: fused streaming kernel ----------
__global__ __launch_bounds__(256)
void k2_main(const float* __restrict__ K, const float* __restrict__ V,
             const float* __restrict__ eps, const unsigned char* __restrict__ padding,
             float* __restrict__ ws, float* __restrict__ out) {
  extern __shared__ char smem[];
  unsigned short* kbuf = (unsigned short*)smem;            // [128][KSTR]
  unsigned short* vbuf = kbuf + CHUNK * KSTR;              // [256][VSTR] (V transposed)
  unsigned short* wbuf = vbuf + DIM * VSTR;                // [16][128]   (w transposed, M-pad)
  float* ebuf = (float*)(wbuf + QT_ROWS * CHUNK);          // [8][128]
  float* hsum = ebuf + NH * CHUNK;                         // [8]

  const int b  = blockIdx.y;
  const int l0 = blockIdx.x * CHUNK;
  const int t  = threadIdx.x;
  const int wave = t >> 5, lane = t & 31;
  const int lo = lane & 15, hi = lane >> 4;

  // ---- stage K (bf16 row-major) + V (bf16 transposed) with NT hint, eps ----
  const vf4* Kb = (const vf4*)(K + ((size_t)b * LSEQ + l0) * DIM);
  const vf4* Vb = (const vf4*)(V + ((size_t)b * LSEQ + l0) * DIM);
  for (int i = t; i < CHUNK * (DIM / 4); i += 256) {
    int row = i >> 6, c4 = i & 63;
    vf4 f = __builtin_nontemporal_load(&Kb[row * (DIM / 4) + c4]);
    unsigned int p0 = f32_to_bf16_rn(f.x) | ((unsigned int)f32_to_bf16_rn(f.y) << 16);
    unsigned int p1 = f32_to_bf16_rn(f.z) | ((unsigned int)f32_to_bf16_rn(f.w) << 16);
    *(uint2*)&kbuf[row * KSTR + c4 * 4] = make_uint2(p0, p1);
    vf4 g = __builtin_nontemporal_load(&Vb[row * (DIM / 4) + c4]);
    int c = c4 * 4;
    vbuf[(c + 0) * VSTR + row] = f32_to_bf16_rn(g.x);
    vbuf[(c + 1) * VSTR + row] = f32_to_bf16_rn(g.y);
    vbuf[(c + 2) * VSTR + row] = f32_to_bf16_rn(g.z);
    vbuf[(c + 3) * VSTR + row] = f32_to_bf16_rn(g.w);
  }
  const float* Eb = eps + (size_t)b * NH * LSEQ + l0;
  for (int i = t; i < NH * CHUNK; i += 256) {
    int h = i >> 7, r = i & 127;
    ebuf[i] = __builtin_nontemporal_load(&Eb[(size_t)h * LSEQ + r]);
  }
  for (int i = t; i < NH * CHUNK; i += 256)     // zero M-pad rows 8..15 of wbuf
    wbuf[NH * CHUNK + i] = 0;
  if (t < NH) hsum[t] = 0.0f;
  __syncthreads();

  // ---- phase A: mu tile = K_chunk . qt^T  (M=16 rows/wave, N=16(8 heads), K=256) ----
  const unsigned short* qtb =
      (const unsigned short*)(ws + WS_QT) + (size_t)b * QT_ROWS * DIM;
  const unsigned short* qrow = qtb + lo * DIM;   // rows 8..15 are zero-padded
  v8f acc = {0.f, 0.f, 0.f, 0.f, 0.f, 0.f, 0.f, 0.f};
  #pragma unroll
  for (int s = 0; s < 8; ++s) {        // 8 x k32 steps over D=256
    FragBF a, bm;
    int row = wave * 16 + lo;
    int ab = row * KSTR + s * 32 + hi * 16;
    a.q[0] = *(const uint4*)&kbuf[ab];
    a.q[1] = *(const uint4*)&kbuf[ab + 8];
    const unsigned short* p = qrow + s * 32 + hi * 16;
    bm.q[0] = *(const uint4*)p;
    bm.q[1] = *(const uint4*)(p + 8);
    acc = __builtin_amdgcn_wmma_f32_16x16x32_bf16(false, a.v, false, bm.v,
                                                  (short)0, acc, false, false);
  }

  // ---- epilogue: mu out (NT), w = exp(mu+eps) masked, head sums, w -> LDS ----
  float* mu_out = out + BSZ * DIM;     // mu region of concatenated output
  const unsigned char* padb = padding + (size_t)b * LSEQ + l0;
  if (lo < NH) {
    const int h = lo;
    const float cb = ws[WS_C + b * NH + h];
    float sloc = 0.0f;
    #pragma unroll
    for (int j = 0; j < 8; ++j) {
      int r = wave * 16 + hi * 8 + j;
      float mu = acc[j] + cb;
      __builtin_nontemporal_store(mu, &mu_out[(size_t)(b * NH + h) * LSEQ + l0 + r]);
      float w = padb[r] ? 0.0f : __expf(mu + ebuf[h * CHUNK + r]);
      sloc += w;
      wbuf[h * CHUNK + r] = f32_to_bf16_rn(w);
    }
    __hip_atomic_fetch_add(&hsum[h], sloc, __ATOMIC_RELAXED, __HIP_MEMORY_SCOPE_WORKGROUP);
  }
  __syncthreads();

  // ---- phase B: U_part = w^T . V_chunk  (M=16(8 heads), K=128, N=256) ----
  float* Ub = ws + WS_U + (size_t)b * NH * DIM;
  #pragma unroll
  for (int tile = 0; tile < 2; ++tile) {
    const int c0 = wave * 32 + tile * 16;
    v8f acc2 = {0.f, 0.f, 0.f, 0.f, 0.f, 0.f, 0.f, 0.f};
    #pragma unroll
    for (int ks = 0; ks < 4; ++ks) {   // 4 x k32 steps over CHUNK=128
      FragBF a2, b2;
      int ab = lo * CHUNK + ks * 32 + hi * 16;     // rows 8..15 zero-padded in LDS
      a2.q[0] = *(const uint4*)&wbuf[ab];
      a2.q[1] = *(const uint4*)&wbuf[ab + 8];
      int col = c0 + lo;
      int vb = col * VSTR + ks * 32 + hi * 16;
      b2.q[0] = *(const uint4*)&vbuf[vb];
      b2.q[1] = *(const uint4*)&vbuf[vb + 8];
      acc2 = __builtin_amdgcn_wmma_f32_16x16x32_bf16(false, a2.v, false, b2.v,
                                                     (short)0, acc2, false, false);
    }
    if (hi == 0) {                     // lanes 0-15 hold heads 0-7 in VGPRs 0-7
      int col = c0 + lo;
      #pragma unroll
      for (int j = 0; j < 8; ++j)
        __hip_atomic_fetch_add(&Ub[j * DIM + col], acc2[j],
                               __ATOMIC_RELAXED, __HIP_MEMORY_SCOPE_AGENT);
    }
  }
  if (t < NH)
    __hip_atomic_fetch_add(&ws[WS_DEN + b * NH + t], hsum[t],
                           __ATOMIC_RELAXED, __HIP_MEMORY_SCOPE_AGENT);
}

// ---------- k3: ctx = (U@Wv_h.T + S*bv)/(S+eps); @Wo.T + bo; LN; +Q ----------
__global__ void k3_epi(const float* __restrict__ Q, const float* __restrict__ Wv,
                       const float* __restrict__ bv, const float* __restrict__ Wo,
                       const float* __restrict__ bo, const float* __restrict__ ln_g,
                       const float* __restrict__ ln_b, const float* __restrict__ ws,
                       float* __restrict__ out) {
  __shared__ float cbuf[DIM];
  __shared__ float ybuf[DIM];
  __shared__ float stats[2];
  int b = blockIdx.x, j = threadIdx.x;
  int h = j >> 5;
  float S = ws[WS_DEN + b * NH + h];
  const float* u  = ws + WS_U + (size_t)(b * NH + h) * DIM;
  const float* wv = Wv + (size_t)j * DIM;
  float a = 0.0f;
  #pragma unroll 4
  for (int k = 0; k < DIM; ++k) a += u[k] * wv[k];
  cbuf[j] = (a + S * bv[j]) / (S + 1e-9f);
  __syncthreads();
  const float* wo = Wo + (size_t)j * DIM;
  float y = bo[j];
  #pragma unroll 4
  for (int k = 0; k < DIM; ++k) y += cbuf[k] * wo[k];
  ybuf[j] = y;
  __syncthreads();
  if (j == 0) {
    float m = 0.0f;
    for (int k = 0; k < DIM; ++k) m += ybuf[k];
    m *= (1.0f / DIM);
    float v = 0.0f;
    for (int k = 0; k < DIM; ++k) { float d = ybuf[k] - m; v += d * d; }
    stats[0] = m; stats[1] = v * (1.0f / DIM);
  }
  __syncthreads();
  out[(size_t)b * DIM + j] =
      (y - stats[0]) * rsqrtf(stats[1] + 1e-5f) * ln_g[j] + ln_b[j] + Q[(size_t)b * DIM + j];
}

extern "C" void kernel_launch(void* const* d_in, const int* in_sizes, int n_in,
                              void* d_out, int out_size, void* d_ws, size_t ws_size,
                              hipStream_t stream) {
  const float* Q    = (const float*)d_in[0];
  const float* K    = (const float*)d_in[1];
  const float* V    = (const float*)d_in[2];
  const float* eps  = (const float*)d_in[3];
  const unsigned char* pad = (const unsigned char*)d_in[4];
  const float* Wq   = (const float*)d_in[5];
  const float* bq   = (const float*)d_in[6];
  const float* Wk   = (const float*)d_in[7];
  const float* bk   = (const float*)d_in[8];
  const float* Wv   = (const float*)d_in[9];
  const float* bv   = (const float*)d_in[10];
  const float* Wo   = (const float*)d_in[11];
  const float* bo   = (const float*)d_in[12];
  const float* ln_g = (const float*)d_in[13];
  const float* ln_b = (const float*)d_in[14];
  float* ws  = (float*)d_ws;
  float* out = (float*)d_out;

  k0_prep<<<BSZ, DIM, 0, stream>>>(Q, Wq, bq, ws);
  k1_qt  <<<BSZ, DIM, 0, stream>>>(Wk, bk, ws);
  size_t smem = (size_t)(CHUNK * KSTR + DIM * VSTR + QT_ROWS * CHUNK) * 2
              + (size_t)(NH * CHUNK + NH) * 4;
  k2_main<<<dim3(NCHUNK, BSZ), 256, smem, stream>>>(K, V, eps, pad, ws, out);
  k3_epi <<<BSZ, DIM, 0, stream>>>(Q, Wv, bv, Wo, bo, ln_g, ln_b, ws, out);
}
